// SelfAttention_28810640621923
// MI455X (gfx1250) — matrix operand order repaired and checked by reading, not compiled, not run on hardware
//
#include <hip/hip_runtime.h>
#include <hip/hip_bf16.h>

// ---- CDNA5 WMMA types ----
typedef __attribute__((ext_vector_type(16))) __bf16 v16bf;
typedef __attribute__((ext_vector_type(8)))  __bf16 v8bf;
typedef __attribute__((ext_vector_type(4)))  __bf16 v4bf;
typedef __attribute__((ext_vector_type(8)))  float  v8f;
typedef __attribute__((ext_vector_type(4)))  float  v4f;

#define CIN   1024
#define C3    3072
#define NTOK  2048
#define NHEAD 16
#define DH    64
#define MROWS 8192                   // 2 streams * B(2) * N(2048)
#define HSZ   (64 * NTOK * DH)       // elements per Q/K/VT/O buffer (64 head-problems)

__device__ __forceinline__ v8f wmma_bf16(v16bf a, v16bf b, v8f c) {
    return __builtin_amdgcn_wmma_f32_16x16x32_bf16(false, a, false, b, (short)0, c, false, false);
}

__device__ __forceinline__ v16bf cat8(v8bf lo, v8bf hi) {
    return __builtin_shufflevector(lo, hi, 0,1,2,3,4,5,6,7,8,9,10,11,12,13,14,15);
}

// =====================================================================
// Kernel 0a: fp32 [x;y] -> bf16 XY (row-major [8192][1024])
// =====================================================================
__global__ void __launch_bounds__(256)
cvt_xy_kernel(const float* __restrict__ x, const float* __restrict__ y,
              __bf16* __restrict__ XY)
{
    const size_t i = ((size_t)blockIdx.x * 256 + threadIdx.x) * 4;
    const size_t half = (size_t)4096 * CIN;
    const float* src = (i < half) ? (x + i) : (y + (i - half));
    v4f f = *(const v4f*)src;
    v4bf b;
    #pragma unroll
    for (int q = 0; q < 4; ++q) b[q] = (__bf16)f[q];
    *(v4bf*)(XY + i) = b;
}

// =====================================================================
// Kernel 0b: transpose-convert  W[K][N] fp32 -> WT[N][K] bf16
// (LDS 64x64 tile; coalesced float4 reads, 8B vector writes)
// =====================================================================
__global__ void __launch_bounds__(256)
tcvt_kernel(const float* __restrict__ W, __bf16* __restrict__ WT, int K, int N)
{
    __shared__ __bf16 tile[64][72];
    const int k0 = blockIdx.y * 64;
    const int n0 = blockIdx.x * 64;
    const int rg = threadIdx.x >> 4;          // 0..15
    const int cg = (threadIdx.x & 15) * 4;    // 0,4,..,60
    #pragma unroll
    for (int rr = 0; rr < 4; ++rr) {
        const int row = rg * 4 + rr;
        v4f f = *(const v4f*)(W + (size_t)(k0 + row) * N + n0 + cg);
        #pragma unroll
        for (int q = 0; q < 4; ++q) tile[row][cg + q] = (__bf16)f[q];
    }
    __syncthreads();
    #pragma unroll
    for (int rr = 0; rr < 4; ++rr) {
        const int nrow = rg * 4 + rr;
        v4bf b;
        #pragma unroll
        for (int q = 0; q < 4; ++q) b[q] = tile[cg + q][nrow];
        *(v4bf*)(WT + (size_t)(n0 + nrow) * K + k0 + cg) = b;
    }
}

// =====================================================================
// Kernel 1: fused QKV GEMM, both streams. A = XY bf16, B = WQT bf16
// (transposed -> contiguous 32B B-fragments). Wave tile 32x64:
// 8 v_wmma_f32_16x16x32_bf16 per K-step, 8 vector loads.
// Epilogue scatters Q (pre-scaled by D^-0.5*log2e), K row-major,
// V transposed [D][N], all bf16.
// =====================================================================
__global__ void __launch_bounds__(256)
qkv_kernel(const __bf16* __restrict__ XY, const __bf16* __restrict__ WQT,
           __bf16* __restrict__ Qb, __bf16* __restrict__ Kb, __bf16* __restrict__ VTb)
{
    const int lane = threadIdx.x & 31;
    const int widx = threadIdx.x >> 5;
    const int mt = blockIdx.y * 8 + widx;   // 0..255 (32-row tiles)
    const int nt = blockIdx.x;              // 0..47  (64-col tiles)
    const int ln = lane & 15, hh = lane >> 4;

    const int m0 = mt * 32;
    const int n0 = nt * 64;

    v8f acc[2][4] = {};
    for (int k0 = 0; k0 < CIN; k0 += 32) {
        v16bf af[2];
        #pragma unroll
        for (int ti = 0; ti < 2; ++ti) {
            const __bf16* p = XY + (size_t)(m0 + ti * 16 + ln) * CIN + k0 + hh * 16;
            af[ti] = cat8(*(const v8bf*)p, *(const v8bf*)(p + 8));
        }
        v16bf bm[4];
        #pragma unroll
        for (int tj = 0; tj < 4; ++tj) {
            const __bf16* bp = WQT + (size_t)(n0 + tj * 16 + ln) * CIN + k0 + hh * 16;
            bm[tj] = *(const v16bf*)bp;     // contraction contiguous: 32B load
        }
        #pragma unroll
        for (int ti = 0; ti < 2; ++ti)
            #pragma unroll
            for (int tj = 0; tj < 4; ++tj)
                acc[ti][tj] = wmma_bf16(af[ti], bm[tj], acc[ti][tj]);
    }

    const float QSCALE = 0.125f * 1.44269504089f;  // D^-0.5 * log2(e)
    #pragma unroll
    for (int ti = 0; ti < 2; ++ti) {
        #pragma unroll
        for (int tj = 0; tj < 4; ++tj) {
            const int c0  = n0 + tj * 16;          // 16-col tile: sel/h uniform
            const int sel = c0 >> 10;              // 0=Q 1=K 2=V
            const int h   = (c0 & 1023) >> 6;
            const int d   = (c0 & 63) + ln;
            const int rb  = m0 + ti * 16 + hh * 8; // + j (rows contiguous)
            const int s   = rb >> 12;
            const int bb  = (rb >> 11) & 1;
            const int n   = rb & 2047;
            const int g   = (s * 2 + bb) * NHEAD + h;
            if (sel == 0) {
                __bf16* q = Qb + (size_t)g * NTOK * DH + (size_t)n * DH + d;
                #pragma unroll
                for (int j = 0; j < 8; ++j) q[(size_t)j * DH] = (__bf16)(acc[ti][tj][j] * QSCALE);
            } else if (sel == 1) {
                __bf16* kk = Kb + (size_t)g * NTOK * DH + (size_t)n * DH + d;
                #pragma unroll
                for (int j = 0; j < 8; ++j) kk[(size_t)j * DH] = (__bf16)acc[ti][tj][j];
            } else {
                __bf16* vt = VTb + (size_t)g * NTOK * DH + (size_t)d * NTOK + n;
                v8bf vv;
                #pragma unroll
                for (int j = 0; j < 8; ++j) vv[j] = (__bf16)acc[ti][tj][j];
                *(v8bf*)vt = vv;                   // 8 contiguous tokens, 16B aligned
            }
        }
    }
}

// =====================================================================
// Kernel 2: flash attention. One wave = one 16-row Q tile of one head.
// KV streamed in 32-token steps: 4 WMMA for S, shfl_xor online softmax,
// P staged via per-wave LDS to A-layout, 4 WMMA for O += P*V.
// Value swap: stream s attends over V of stream 1-s  (g ^ 32).
// =====================================================================
__global__ void __launch_bounds__(256)
attn_kernel(const __bf16* __restrict__ Qb, const __bf16* __restrict__ Kb,
            const __bf16* __restrict__ VTb, __bf16* __restrict__ Ob)
{
    __shared__ __bf16 plds[8][16 * 32];
    const int lane = threadIdx.x & 31;
    const int widx = threadIdx.x >> 5;
    const int g  = blockIdx.y;                 // head-problem 0..63 (s*32+bb*16+h)
    const int qt = blockIdx.x * 8 + widx;      // q tile 0..127
    const int ln = lane & 15, hh = lane >> 4;

    const __bf16* Q  = Qb  + (size_t)g * NTOK * DH;
    const __bf16* K  = Kb  + (size_t)g * NTOK * DH;
    const __bf16* VT = VTb + (size_t)(g ^ 32) * NTOK * DH;   // swapped stream

    const int q0 = qt * 16;
    v16bf aq[2];
    #pragma unroll
    for (int c = 0; c < 2; ++c) {
        const __bf16* p = Q + (size_t)(q0 + ln) * DH + c * 32 + hh * 16;
        aq[c] = cat8(*(const v8bf*)p, *(const v8bf*)(p + 8));
    }

    v8f o[4] = {};
    float mrow[8], lrow[8];
    #pragma unroll
    for (int j = 0; j < 8; ++j) { mrow[j] = -1e30f; lrow[j] = 0.0f; }

    __bf16* pl = plds[widx];

    for (int kv0 = 0; kv0 < NTOK; kv0 += 32) {
        v8f s0 = {}, s1 = {};
        #pragma unroll
        for (int c = 0; c < 2; ++c) {
            const __bf16* kp0 = K + (size_t)(kv0 + ln) * DH + c * 32 + hh * 16;
            s0 = wmma_bf16(aq[c], *(const v16bf*)kp0, s0);
            const __bf16* kp1 = K + (size_t)(kv0 + 16 + ln) * DH + c * 32 + hh * 16;
            s1 = wmma_bf16(aq[c], *(const v16bf*)kp1, s1);
        }
        // online softmax, row = j + 8*hh lives in a contiguous 16-lane group
        #pragma unroll
        for (int j = 0; j < 8; ++j) {
            float v = fmaxf(s0[j], s1[j]);
            #pragma unroll
            for (int off = 1; off < 16; off <<= 1) v = fmaxf(v, __shfl_xor(v, off, 32));
            const float mn = fmaxf(mrow[j], v);
            const float al = exp2f(mrow[j] - mn);
            const float p0 = exp2f(s0[j] - mn);
            const float p1 = exp2f(s1[j] - mn);
            float rs = p0 + p1;
            #pragma unroll
            for (int off = 1; off < 16; off <<= 1) rs += __shfl_xor(rs, off, 32);
            lrow[j] = lrow[j] * al + rs;
            mrow[j] = mn;
            pl[(j + hh * 8) * 32 + ln]      = (__bf16)p0;
            pl[(j + hh * 8) * 32 + 16 + ln] = (__bf16)p1;
            #pragma unroll
            for (int t = 0; t < 4; ++t) o[t][j] *= al;
        }
        // P (C layout in LDS) -> A-layout fragment
        const __bf16* pp = pl + ln * 32 + hh * 16;
        v16bf ap = cat8(*(const v8bf*)pp, *(const v8bf*)(pp + 8));
        #pragma unroll
        for (int t = 0; t < 4; ++t) {
            const __bf16* vp = VT + (size_t)(t * 16 + ln) * NTOK + kv0 + hh * 16;
            o[t] = wmma_bf16(ap, *(const v16bf*)vp, o[t]);   // 32B contiguous B frag
        }
    }

    const int s  = g >> 5;
    const int bb = (g >> 4) & 1;
    const int h  = g & 15;
    const size_t rowbase = ((size_t)s * 4096 + bb * 2048 + q0 + hh * 8) * 1024 + h * 64;
    #pragma unroll
    for (int t = 0; t < 4; ++t) {
        __bf16* op = Ob + rowbase + t * 16 + ln;
        #pragma unroll
        for (int j = 0; j < 8; ++j) op[(size_t)j * 1024] = (__bf16)(o[t][j] / lrow[j]);
    }
}

// =====================================================================
// Kernel 3: out = O @ w_proj + b_proj + residual (fp32 out, concat layout)
// A = Ob bf16, B = WPT bf16 (transposed). Wave tile 32x64.
// =====================================================================
__global__ void __launch_bounds__(256)
proj_kernel(const __bf16* __restrict__ Ob, const __bf16* __restrict__ WPT,
            const float* __restrict__ bp, const float* __restrict__ x,
            const float* __restrict__ y, float* __restrict__ out)
{
    const int lane = threadIdx.x & 31;
    const int widx = threadIdx.x >> 5;
    const int mt = blockIdx.y * 8 + widx;   // 0..255
    const int nt = blockIdx.x;              // 0..15
    const int ln = lane & 15, hh = lane >> 4;

    const int m0 = mt * 32;
    const int n0 = nt * 64;
    const __bf16* A = Ob + (size_t)m0 * CIN;
    const float* R = (m0 < 4096) ? (x + (size_t)m0 * CIN)
                                 : (y + (size_t)(m0 - 4096) * CIN);

    v8f acc[2][4] = {};
    for (int k0 = 0; k0 < CIN; k0 += 32) {
        v16bf af[2];
        #pragma unroll
        for (int ti = 0; ti < 2; ++ti) {
            const __bf16* p = A + (size_t)(ti * 16 + ln) * CIN + k0 + hh * 16;
            af[ti] = cat8(*(const v8bf*)p, *(const v8bf*)(p + 8));
        }
        v16bf bm[4];
        #pragma unroll
        for (int tj = 0; tj < 4; ++tj) {
            const __bf16* bptr = WPT + (size_t)(n0 + tj * 16 + ln) * CIN + k0 + hh * 16;
            bm[tj] = *(const v16bf*)bptr;   // contiguous 32B B fragment
        }
        #pragma unroll
        for (int ti = 0; ti < 2; ++ti)
            #pragma unroll
            for (int tj = 0; tj < 4; ++tj)
                acc[ti][tj] = wmma_bf16(af[ti], bm[tj], acc[ti][tj]);
    }

    #pragma unroll
    for (int ti = 0; ti < 2; ++ti) {
        #pragma unroll
        for (int tj = 0; tj < 4; ++tj) {
            const int col = n0 + tj * 16 + ln;
            const int lr  = ti * 16 + hh * 8;        // row within 32-row tile (+j)
            const float bias = bp[col];
            #pragma unroll
            for (int j = 0; j < 8; ++j) {
                const size_t idx = (size_t)(m0 + lr + j) * CIN + col;  // concat layout == row order
                out[idx] = acc[ti][tj][j] + bias + R[(size_t)(lr + j) * CIN + col];
            }
        }
    }
}

extern "C" void kernel_launch(void* const* d_in, const int* in_sizes, int n_in,
                              void* d_out, int out_size, void* d_ws, size_t ws_size,
                              hipStream_t stream) {
    const float* x     = (const float*)d_in[0];
    const float* y     = (const float*)d_in[1];
    const float* wqkv  = (const float*)d_in[2];
    const float* wproj = (const float*)d_in[3];
    const float* bproj = (const float*)d_in[4];
    float* out = (float*)d_out;

    __bf16* Qb  = (__bf16*)d_ws;
    __bf16* Kb  = Qb  + HSZ;
    __bf16* VTb = Kb  + HSZ;
    __bf16* XY  = VTb + HSZ;                       // [8192][1024] bf16
    __bf16* Ob  = XY;                              // reuse: XY dead after qkv_kernel
    __bf16* WQT = XY  + (size_t)MROWS * CIN;       // [3072][1024] bf16
    __bf16* WPT = WQT + (size_t)C3 * CIN;          // [1024][1024] bf16

    cvt_xy_kernel<<<dim3(MROWS * CIN / 1024), 256, 0, stream>>>(x, y, XY);
    tcvt_kernel  <<<dim3(C3 / 64,  CIN / 64), 256, 0, stream>>>(wqkv,  WQT, CIN, C3);
    tcvt_kernel  <<<dim3(CIN / 64, CIN / 64), 256, 0, stream>>>(wproj, WPT, CIN, CIN);

    qkv_kernel <<<dim3(48, 32), 256, 0, stream>>>(XY, WQT, Qb, Kb, VTb);
    attn_kernel<<<dim3(16, 64), 256, 0, stream>>>(Qb, Kb, VTb, Ob);
    proj_kernel<<<dim3(16, 32), 256, 0, stream>>>(Ob, WPT, bproj, x, y, out);
}
